// MultiCenterLoss_90409061580855
// MI455X (gfx1250) — matrix operand order repaired and checked by reading, not compiled, not run on hardware
//
#include <hip/hip_runtime.h>

#define N_SAMPLES 16384
#define D_DIM     1024
#define C_CENTERS 1024
#define EPS       1e-6f
#define DENOM_EPS 1e-5f

typedef __bf16 bf16;
typedef __attribute__((ext_vector_type(16))) __bf16 v16bf;
typedef __attribute__((ext_vector_type(8)))  float  v8f;
typedef unsigned int u32x4 __attribute__((ext_vector_type(4)));
typedef int          i32x8 __attribute__((ext_vector_type(8)));
typedef int          i32x4 __attribute__((ext_vector_type(4)));

union FragA { uint4 q[2]; v16bf v; };

// ---------------------------------------------------------------------------
// Prologue: per-row (sum, sumsq) -> term[row], plus bf16 hi/lo split.
//   term = sumsq + sign2eps*sum + addc
// One block (256 thr) per row of 1024 f32.
// ---------------------------------------------------------------------------
__global__ __launch_bounds__(256)
void rowprep(const float* __restrict__ src,
             unsigned short* __restrict__ hi,
             unsigned short* __restrict__ lo,
             float* __restrict__ term,
             float sign2eps, float addc)
{
    __shared__ float s1[8], s2[8];
    const int row = blockIdx.x;
    const int tid = threadIdx.x;

    const float4 x = ((const float4*)(src + (size_t)row * D_DIM))[tid];
    float vals[4] = {x.x, x.y, x.z, x.w};

    union { unsigned short s[4]; uint2 u; } ph, pl;
    float sum = 0.f, sq = 0.f;
#pragma unroll
    for (int i = 0; i < 4; ++i) {
        float v = vals[i];
        sum += v;
        sq  += v * v;
        __bf16 h  = (__bf16)v;
        float  hf = (float)h;
        __bf16 l  = (__bf16)(v - hf);
        ph.s[i] = __builtin_bit_cast(unsigned short, h);
        pl.s[i] = __builtin_bit_cast(unsigned short, l);
    }
    ((uint2*)(hi + (size_t)row * D_DIM))[tid] = ph.u;
    ((uint2*)(lo + (size_t)row * D_DIM))[tid] = pl.u;

#pragma unroll
    for (int off = 16; off > 0; off >>= 1) {
        sum += __shfl_down(sum, off);
        sq  += __shfl_down(sq,  off);
    }
    const int w = tid >> 5, lane = tid & 31;
    if (lane == 0) { s1[w] = sum; s2[w] = sq; }
    __syncthreads();
    if (tid == 0) {
        float S = 0.f, Q = 0.f;
#pragma unroll
        for (int i = 0; i < 8; ++i) { S += s1[i]; Q += s2[i]; }
        term[row] = Q + sign2eps * S + addc;
    }
}

// ---------------------------------------------------------------------------
// Build a TDM D# for a 2D tile load: tile_d0 elems (2B) x tile_d1 rows,
// row stride = stride elems, from global 'ga' into LDS byte offset 'lds_off'.
// Field packing per CDNA5 ISA.
// Toolchain here exposes the 6-arg builtin (clang-23 / therock-10.0 form):
//   (u32x4 g0, i32x8 g1, i32x4 g2, i32x4 g3, i32x8 extra, i32 cpol)
// ---------------------------------------------------------------------------
__device__ __forceinline__
void tdm_load_2d(unsigned lds_off, unsigned long long ga,
                 unsigned tile_d0, unsigned tile_d1, unsigned stride)
{
    u32x4 g0;
    g0[0] = 1u;                                   // count=1, user descriptor
    g0[1] = lds_off;                              // lds_addr (bytes)
    g0[2] = (unsigned)(ga & 0xFFFFFFFFull);       // global_addr[31:0]
    g0[3] = (unsigned)((ga >> 32) & 0x1FFFFFFull) // global_addr[56:32]
          | (2u << 30);                           // type=2 ("image")

    i32x8 g1;
    g1[0] = (int)(1u << 16);                      // data_size=1 -> 2 bytes
    g1[1] = (int)(tile_d0 << 16);                 // tensor_dim0[15:0]  (== tile span)
    g1[2] = (int)(0xFFFFu << 16);                 // tensor_dim1[15:0]  (large: no OOB clip)
    g1[3] = (int)(tile_d0 << 16);                 // tile_dim0 in [127:112]
    g1[4] = (int)tile_d1;                         // tile_dim1 in [143:128]; tile_dim2=0
    g1[5] = (int)stride;                          // tensor_dim0_stride[31:0]
    g1[6] = 0;                                    // stride hi / dim1_stride lo
    g1[7] = 0;                                    // dim1_stride hi

    i32x4 g2;                                     // dims 2/3: present but unused
    g2[0] = 1;                                    // tensor_dim2 = 1
    g2[1] = 1;                                    // tensor_dim3 = 1
    g2[2] = 0;                                    // tensor_dim2_stride lo
    g2[3] = 0;                                    // tile_dim3 = 0 (unused)

    i32x4 g3;
    g3[0] = 0;                                    // tensor_dim3_stride lo
    g3[1] = (int)(1u << 16);                      // tensor_dim4[15:0] = 1
    g3[2] = 0;                                    // tile_dim4 = 0 (unused)
    g3[3] = 0;

    i32x8 gx = {0, 0, 0, 0, 0, 0, 0, 0};         // extra operand group: zero-filled

    __builtin_amdgcn_tensor_load_to_lds(g0, g1, g2, g3, gx, 0);
}

// ---------------------------------------------------------------------------
// Main: per block 16 rows x all 1024 centers.
// A hi/lo row-set (64 KB) staged ONCE into LDS by wave 0 via the Tensor Data
// Mover; all 8 waves then read A fragments with ds_load_b128.
// 8 waves; wave w owns centers [w*128, w*128+128) as 4 passes of TWO 16x16
// column tiles (A fragments reused across both accumulators).
// fc via bf16-split 3-WMMA emulation of fp32; min tracked on (cterm - 2*fc),
// fterm/clamp/sqrt applied once per row at the end.
// ---------------------------------------------------------------------------
__global__ __launch_bounds__(256)
void minloss_gemm(const unsigned short* __restrict__ fh,
                  const unsigned short* __restrict__ fl,
                  const unsigned short* __restrict__ ch,
                  const unsigned short* __restrict__ cl,
                  const float* __restrict__ fterm,
                  const float* __restrict__ cterm,
                  float* __restrict__ min_out)
{
    __shared__ unsigned short ldsA[2][16 * D_DIM];   // [hi/lo][row*D + k], 64 KB
    __shared__ float smem[8][16];

    const int tid  = threadIdx.x;
    const int w    = tid >> 5;
    const int lane = tid & 31;
    const int half = lane >> 4;
    const int r    = lane & 15;

    const int m_base = blockIdx.x * 16;

    // Generic LDS pointer: low 32 bits are the wave-relative LDS byte offset.
    const unsigned ldsA_hi_off = (unsigned)(unsigned long long)(const void*)&ldsA[0][0];
    const unsigned ldsA_lo_off = (unsigned)(unsigned long long)(const void*)&ldsA[1][0];

    if (w == 0) {
        const unsigned long long gah =
            (unsigned long long)(const void*)(fh + (size_t)m_base * D_DIM);
        const unsigned long long gal =
            (unsigned long long)(const void*)(fl + (size_t)m_base * D_DIM);
        tdm_load_2d(ldsA_hi_off, gah, D_DIM, 16, D_DIM);
        tdm_load_2d(ldsA_lo_off, gal, D_DIM, 16, D_DIM);
        __builtin_amdgcn_s_wait_tensorcnt(0);
    }
    __syncthreads();   // A tiles resident for the whole block

    const unsigned short* pAh = &ldsA[0][r * D_DIM];
    const unsigned short* pAl = &ldsA[1][r * D_DIM];

    float rminsq[8];
#pragma unroll
    for (int v = 0; v < 8; ++v) rminsq[v] = 3.4e38f;

    for (int ct = 0; ct < 4; ++ct) {
        const int c0 = w * 128 + ct * 32;
        const size_t rowB0 = (size_t)(c0 + r) * D_DIM + 16 * half;
        const size_t rowB1 = (size_t)(c0 + 16 + r) * D_DIM + 16 * half;
        const unsigned short* pB0h = ch + rowB0;
        const unsigned short* pB0l = cl + rowB0;
        const unsigned short* pB1h = ch + rowB1;
        const unsigned short* pB1l = cl + rowB1;
        const float ct0 = cterm[c0 + r];
        const float ct1 = cterm[c0 + 16 + r];

        v8f acc0 = {};
        v8f acc1 = {};
#pragma unroll 2
        for (int k0 = 0; k0 < D_DIM; k0 += 32) {
            // k order used here for A and B: lane half h holds contiguous K = 16h..16h+15
            FragA ah, al;
            ah.q[0] = *(const uint4*)(pAh + k0 + 16 * half);       // ds_load_b128
            ah.q[1] = *(const uint4*)(pAh + k0 + 16 * half + 8);
            al.q[0] = *(const uint4*)(pAl + k0 + 16 * half);
            al.q[1] = *(const uint4*)(pAl + k0 + 16 * half + 8);
            v16bf b0h = *(const v16bf*)(pB0h + k0);                // global b128 x2
            v16bf b0l = *(const v16bf*)(pB0l + k0);
            v16bf b1h = *(const v16bf*)(pB1h + k0);
            v16bf b1l = *(const v16bf*)(pB1l + k0);

            // fp32-emulation: f.c ~= fh.ch + fh.cl + fl.ch (f32 accumulate)
            acc0 = __builtin_amdgcn_wmma_f32_16x16x32_bf16(false, ah.v, false, b0h,
                                                           (short)0, acc0, false, false);
            acc1 = __builtin_amdgcn_wmma_f32_16x16x32_bf16(false, ah.v, false, b1h,
                                                           (short)0, acc1, false, false);
            acc0 = __builtin_amdgcn_wmma_f32_16x16x32_bf16(false, ah.v, false, b0l,
                                                           (short)0, acc0, false, false);
            acc1 = __builtin_amdgcn_wmma_f32_16x16x32_bf16(false, ah.v, false, b1l,
                                                           (short)0, acc1, false, false);
            acc0 = __builtin_amdgcn_wmma_f32_16x16x32_bf16(false, al.v, false, b0h,
                                                           (short)0, acc0, false, false);
            acc1 = __builtin_amdgcn_wmma_f32_16x16x32_bf16(false, al.v, false, b1h,
                                                           (short)0, acc1, false, false);
        }

#pragma unroll
        for (int v = 0; v < 8; ++v) {
            float t0 = __builtin_fmaf(-2.0f, acc0[v], ct0);
            float t1 = __builtin_fmaf(-2.0f, acc1[v], ct1);
            rminsq[v] = fminf(rminsq[v], fminf(t0, t1));
        }
    }

    // min over the 16 column-lanes within each half-wave
#pragma unroll
    for (int v = 0; v < 8; ++v) {
        float m = rminsq[v];
#pragma unroll
        for (int off = 1; off < 16; off <<= 1)
            m = fminf(m, __shfl_xor(m, off));
        rminsq[v] = m;
    }
    if (r == 0) {
#pragma unroll
        for (int v = 0; v < 8; ++v)
            smem[w][v + 8 * half] = rminsq[v];   // row index v + 8*half
    }
    __syncthreads();
    if (tid < 16) {
        float m = smem[0][tid];
#pragma unroll
        for (int i = 1; i < 8; ++i) m = fminf(m, smem[i][tid]);
        float sq = fterm[m_base + tid] + m;      // add row constant after min
        min_out[m_base + tid] = __builtin_amdgcn_sqrtf(fmaxf(sq, 0.0f));
    }
}

// ---------------------------------------------------------------------------
// Epilogue: loss = sum(min_dist * (label==0)) / (count + 1e-5)
// ---------------------------------------------------------------------------
__global__ __launch_bounds__(256)
void final_loss(const float* __restrict__ min_dist,
                const int* __restrict__ labels,
                float* __restrict__ out)
{
    __shared__ float ssum[256];
    __shared__ float scnt[256];
    const int tid = threadIdx.x;
    float s = 0.f, c = 0.f;
    for (int i = tid; i < N_SAMPLES; i += 256) {
        float m = (labels[i] == 0) ? 1.0f : 0.0f;
        s += min_dist[i] * m;
        c += m;
    }
    ssum[tid] = s; scnt[tid] = c;
    __syncthreads();
    for (int off = 128; off > 0; off >>= 1) {
        if (tid < off) { ssum[tid] += ssum[tid + off]; scnt[tid] += scnt[tid + off]; }
        __syncthreads();
    }
    if (tid == 0) out[0] = ssum[0] / (scnt[0] + DENOM_EPS);
}

// ---------------------------------------------------------------------------
extern "C" void kernel_launch(void* const* d_in, const int* in_sizes, int n_in,
                              void* d_out, int out_size, void* d_ws, size_t ws_size,
                              hipStream_t stream)
{
    const float* features = (const float*)d_in[0];
    const int*   labels   = (const int*)d_in[1];
    const float* centers  = (const float*)d_in[2];

    char* ws = (char*)d_ws;
    // Workspace layout (bytes), ~68.1 MB total, all 16B aligned:
    float*          fterm = (float*)(ws + 0);                       //  64 KB
    float*          cterm = (float*)(ws + 65536);                   //   4 KB
    float*          min_d = (float*)(ws + 69632);                   //  64 KB
    unsigned short* c_hi  = (unsigned short*)(ws + 135168);         //   2 MB
    unsigned short* c_lo  = (unsigned short*)(ws + 135168 + 2097152);
    unsigned short* f_hi  = (unsigned short*)(ws + 135168 + 2u * 2097152u);
    unsigned short* f_lo  = (unsigned short*)(ws + 135168 + 2u * 2097152u + 33554432u);

    rowprep<<<C_CENTERS, 256, 0, stream>>>(centers, c_hi, c_lo, cterm,
                                           -2.0f * EPS, 0.0f);
    rowprep<<<N_SAMPLES, 256, 0, stream>>>(features, f_hi, f_lo, fterm,
                                           2.0f * EPS, EPS * EPS * (float)D_DIM);
    minloss_gemm<<<N_SAMPLES / 16, 256, 0, stream>>>(f_hi, f_lo, c_hi, c_lo,
                                                     fterm, cterm, min_d);
    final_loss<<<1, 256, 0, stream>>>(min_d, labels, (float*)d_out);
}
